// LNPeepholeLSTM_83657372991894
// MI455X (gfx1250) — compile-verified, matrix operand order checked by reading, not run on hardware
//
#include <hip/hip_runtime.h>
#include <math.h>

#define HD   1024
#define BD   64
#define GT   4096          // 4*HD
#define TT   512
#define KSP  4             // K-split partitions (2 per source matrix)
#define EPSF 1e-5f

typedef __attribute__((ext_vector_type(16))) __bf16 v16bf;
typedef __attribute__((ext_vector_type(8)))  float  v8f;

__device__ inline __bf16 to_bf16(float f) {
    unsigned u = __float_as_uint(f);
    unsigned r = (u + 0x7FFFu + ((u >> 16) & 1u)) >> 16;
    unsigned short s = (unsigned short)r;
    return *(__bf16*)&s;
}

__global__ void f32_to_bf16_kernel(const float* __restrict__ in,
                                   __bf16* __restrict__ out, int n) {
    int i = blockIdx.x * blockDim.x + threadIdx.x;
    if (i < n) out[i] = to_bf16(in[i]);
}

// Partial gates: gparts[kp](64 x 4096) = A_kp(64 x 512) @ W_kp^T
//   kp 0,1 -> x_t / W_ih  (K 0..511 / 512..1023)
//   kp 2,3 -> h   / W_hh
// grid: (64 N-strips, 4 K-parts), 128 threads = 4 waves.
// Double-buffered LDS fed by GLOBAL_LOAD_ASYNC_TO_LDS_B128 (ASYNCcnt).
__global__ __launch_bounds__(128) void step_gemm(
    const __bf16* __restrict__ xt,   // 64 x 1024
    const __bf16* __restrict__ hb,   // 64 x 1024
    const __bf16* __restrict__ wih,  // 4096 x 1024 (row-major, K contiguous)
    const __bf16* __restrict__ whh,  // 4096 x 1024
    float* __restrict__ gparts)      // KSP x 64 x 4096
{
    __shared__ __align__(16) __bf16 As[2][64 * 32];
    __shared__ __align__(16) __bf16 Bs[2][64 * 32];

    const int n0   = blockIdx.x * 64;
    const int kp   = blockIdx.y;
    const int tid  = threadIdx.x;
    const int wave = tid >> 5;
    const int lane = tid & 31;
    const int lh   = lane >> 4;   // lane half selects K sub-range
    const int lr   = lane & 15;

    const __bf16* A = (kp < 2) ? xt  : hb;
    const __bf16* W = (kp < 2) ? wih : whh;
    const int kbase = (kp & 1) * 512;

    // issue one 64x32 chunk of A and W into LDS buffer `buf` (async, no VGPRs)
    auto issue = [&](int ch, int buf) {
        const int k0 = kbase + ch * 32;
        for (int i = tid; i < 256; i += 128) {
            int row = i >> 2;
            int kv  = (i & 3) << 3;
            unsigned la = (unsigned)(size_t)&As[buf][row * 32 + kv];
            unsigned lb = (unsigned)(size_t)&Bs[buf][row * 32 + kv];
            unsigned long long ga =
                (unsigned long long)(size_t)&A[row * 1024 + k0 + kv];
            unsigned long long gb =
                (unsigned long long)(size_t)&W[(size_t)(n0 + row) * 1024 + k0 + kv];
            asm volatile("global_load_async_to_lds_b128 %0, %1, off"
                         :: "v"(la), "v"(ga) : "memory");
            asm volatile("global_load_async_to_lds_b128 %0, %1, off"
                         :: "v"(lb), "v"(gb) : "memory");
        }
    };

    v8f acc[4] = {};              // 4 M-tiles x this wave's 16-wide N strip

    const int NCH = 512 / 32;     // 16 chunks
    issue(0, 0);
    for (int ch = 0; ch < NCH; ++ch) {
        const int cur = ch & 1;
        if (ch + 1 < NCH) {
            issue(ch + 1, cur ^ 1);                    // prefetch next chunk
            asm volatile("s_wait_asynccnt 0x4" ::: "memory"); // own prev done
        } else {
            asm volatile("s_wait_asynccnt 0x0" ::: "memory");
        }
        __syncthreads();

        // B fragment (32x16): column = wave's strip col, ISA K interleave
        v16bf bfrag;
        {
            int n = wave * 16 + lr;
            #pragma unroll
            for (int e = 0; e < 16; ++e) {
                int k = e + 8 * lh + ((e & 8) ? 8 : 0);
                bfrag[e] = Bs[cur][n * 32 + k];
            }
        }
        #pragma unroll
        for (int m = 0; m < 4; ++m) {
            v16bf afrag;
            int row = m * 16 + lr;
            #pragma unroll
            for (int e = 0; e < 16; ++e) {
                int k = e + 8 * lh + ((e & 8) ? 8 : 0);
                afrag[e] = As[cur][row * 32 + k];
            }
            acc[m] = __builtin_amdgcn_wmma_f32_16x16x32_bf16(
                false, afrag, false, bfrag, (short)0, acc[m], false, false);
        }
        __syncthreads();
    }

    // C/D layout: M = r + 8*lh (+16*m), N = lr
    float* gout = gparts + (size_t)kp * BD * GT;
    #pragma unroll
    for (int m = 0; m < 4; ++m)
        #pragma unroll
        for (int r = 0; r < 8; ++r) {
            int row = m * 16 + 8 * lh + r;
            gout[(size_t)row * GT + n0 + wave * 16 + lr] = acc[m][r];
        }
}

__device__ inline float block_sum(float v, float* sbuf) {
    #pragma unroll
    for (int o = 16; o > 0; o >>= 1) v += __shfl_down(v, o, 32);
    __syncthreads();
    if ((threadIdx.x & 31) == 0) sbuf[threadIdx.x >> 5] = v;
    __syncthreads();
    if (threadIdx.x == 0) {
        float s = 0.f;
        for (int i = 0; i < 8; ++i) s += sbuf[i];
        sbuf[15] = s;
    }
    __syncthreads();
    return sbuf[15];
}

__device__ inline float sigmoidf_(float x) { return 1.0f / (1.0f + __expf(-x)); }

// one block per batch row (64 blocks x 256 threads, 4 cols/thread)
__global__ __launch_bounds__(256) void step_point(
    const float* __restrict__ gparts, // KSP x 64 x 4096 (f,i,g,o)
    float* __restrict__ c,            // 64 x 1024 (in/out)
    __bf16* __restrict__ hbf,         // 64 x 1024 (out, next-step GEMM operand)
    const float* __restrict__ wch,    // 3H: w_cf, w_ci, w_co
    const float* __restrict__ bias,   // 4H
    const float* __restrict__ gf, const float* __restrict__ gi,
    const float* __restrict__ gg, const float* __restrict__ go,
    const float* __restrict__ gcell, const float* __restrict__ bcell,
    float* __restrict__ out_t,        // 64 x 1024
    float* __restrict__ hf, float* __restrict__ cf, int last)
{
    __shared__ float sbuf[16];
    const int b   = blockIdx.x;
    const int tid = threadIdx.x;
    const float* g0 = gparts + (size_t)b * GT;
    const float* g1 = g0 + (size_t)BD * GT;
    const float* g2 = g1 + (size_t)BD * GT;
    const float* g3 = g2 + (size_t)BD * GT;

    float fin[4], iin[4], gin[4], ohat[4], cv[4];
    float sf = 0, s2f = 0, si = 0, s2i = 0, sg = 0, s2g = 0;
    #pragma unroll
    for (int k = 0; k < 4; ++k) {
        int col = tid + k * 256;
        float cc = c[b * HD + col];
        cv[k] = cc;
        float f = (g0[col] + g1[col] + g2[col] + g3[col])
                  + wch[col] * cc;
        float i = (g0[HD + col] + g1[HD + col] + g2[HD + col] + g3[HD + col])
                  + wch[HD + col] * cc;
        float g = g0[2 * HD + col] + g1[2 * HD + col]
                + g2[2 * HD + col] + g3[2 * HD + col];
        ohat[k] = g0[3 * HD + col] + g1[3 * HD + col]
                + g2[3 * HD + col] + g3[3 * HD + col];
        fin[k] = f; iin[k] = i; gin[k] = g;
        sf += f; s2f += f * f; si += i; s2i += i * i; sg += g; s2g += g * g;
    }
    sf  = block_sum(sf,  sbuf);  s2f = block_sum(s2f, sbuf);
    si  = block_sum(si,  sbuf);  s2i = block_sum(s2i, sbuf);
    sg  = block_sum(sg,  sbuf);  s2g = block_sum(s2g, sbuf);

    const float inv = 1.0f / (float)HD;
    float mf = sf * inv, vf = s2f * inv - mf * mf;
    float mi = si * inv, vi = s2i * inv - mi * mi;
    float mg = sg * inv, vg = s2g * inv - mg * mg;
    float rf = rsqrtf(vf + EPSF), ri = rsqrtf(vi + EPSF), rg = rsqrtf(vg + EPSF);

    float sc = 0, s2c = 0, so = 0, s2o = 0;
    #pragma unroll
    for (int k = 0; k < 4; ++k) {
        int col = tid + k * 256;
        float f = sigmoidf_((fin[k] - mf) * rf * gf[col] + bias[col]);
        float i = sigmoidf_((iin[k] - mi) * ri * gi[col] + bias[HD + col]);
        float g = tanhf((gin[k] - mg) * rg * gg[col] + bias[2 * HD + col]);
        float cn = f * cv[k] + i * g;
        cv[k] = cn;
        float oin = ohat[k] + wch[2 * HD + col] * cn;
        ohat[k] = oin;
        sc += cn; s2c += cn * cn; so += oin; s2o += oin * oin;
    }
    sc = block_sum(sc, sbuf); s2c = block_sum(s2c, sbuf);
    so = block_sum(so, sbuf); s2o = block_sum(s2o, sbuf);

    float mc = sc * inv, vc = s2c * inv - mc * mc;
    float mo = so * inv, vo = s2o * inv - mo * mo;
    float rc = rsqrtf(vc + EPSF), ro = rsqrtf(vo + EPSF);

    #pragma unroll
    for (int k = 0; k < 4; ++k) {
        int col = tid + k * 256;
        float cln = (cv[k] - mc) * rc * gcell[col] + bcell[col];
        float o   = sigmoidf_((ohat[k] - mo) * ro * go[col] + bias[3 * HD + col]);
        float h   = o * tanhf(cln);
        c[b * HD + col]     = cv[k];
        out_t[b * HD + col] = h;
        hbf[b * HD + col]   = to_bf16(h);
        if (last) { hf[b * HD + col] = h; cf[b * HD + col] = cv[k]; }
    }
}

extern "C" void kernel_launch(void* const* d_in, const int* in_sizes, int n_in,
                              void* d_out, int out_size, void* d_ws, size_t ws_size,
                              hipStream_t stream) {
    const float* input  = (const float*)d_in[0];   // 512 x 64 x 1024
    const float* hidden = (const float*)d_in[1];   // 64 x 1024
    const float* cell   = (const float*)d_in[2];   // 64 x 1024
    const float* wih    = (const float*)d_in[3];   // 4096 x 1024
    const float* whh    = (const float*)d_in[4];   // 4096 x 1024
    const float* wch    = (const float*)d_in[5];   // 3072
    const float* bias   = (const float*)d_in[6];   // 4096
    const float* gf     = (const float*)d_in[7];
    const float* gi     = (const float*)d_in[8];
    const float* gg     = (const float*)d_in[9];
    const float* go     = (const float*)d_in[10];
    const float* gcell  = (const float*)d_in[11];
    const float* bcell  = (const float*)d_in[12];
    float* out = (float*)d_out;

    char* ws = (char*)d_ws;
    __bf16* xbf  = (__bf16*)ws;  ws += (size_t)TT * BD * HD * 2;
    __bf16* wihb = (__bf16*)ws;  ws += (size_t)GT * HD * 2;
    __bf16* whhb = (__bf16*)ws;  ws += (size_t)GT * HD * 2;
    __bf16* hbf  = (__bf16*)ws;  ws += (size_t)BD * HD * 2;
    float*  cbuf = (float*)ws;   ws += (size_t)BD * HD * 4;
    float*  gbuf = (float*)ws;   ws += (size_t)KSP * BD * GT * 4;

    int nx = TT * BD * HD;
    f32_to_bf16_kernel<<<(nx + 255) / 256, 256, 0, stream>>>(input, xbf, nx);
    int nw = GT * HD;
    f32_to_bf16_kernel<<<(nw + 255) / 256, 256, 0, stream>>>(wih, wihb, nw);
    f32_to_bf16_kernel<<<(nw + 255) / 256, 256, 0, stream>>>(whh, whhb, nw);
    int nh = BD * HD;
    f32_to_bf16_kernel<<<(nh + 255) / 256, 256, 0, stream>>>(hidden, hbf, nh);
    hipMemcpyAsync(cbuf, cell, (size_t)nh * sizeof(float),
                   hipMemcpyDeviceToDevice, stream);

    float* hf = out + (size_t)TT * BD * HD;
    float* cf = hf + (size_t)BD * HD;

    for (int t = 0; t < TT; ++t) {
        step_gemm<<<dim3(64, KSP), 128, 0, stream>>>(
            xbf + (size_t)t * BD * HD, hbf, wihb, whhb, gbuf);
        step_point<<<64, 256, 0, stream>>>(gbuf, cbuf, hbf, wch, bias,
                                           gf, gi, gg, go, gcell, bcell,
                                           out + (size_t)t * BD * HD,
                                           hf, cf, t == TT - 1 ? 1 : 0);
    }
}
